// GNN_69758858822499
// MI455X (gfx1250) — compile-verified, matrix operand order checked by reading, not run on hardware
//
#include <hip/hip_runtime.h>
#include <hip/hip_bf16.h>
#include <hip/hip_fp16.h>

typedef __attribute__((ext_vector_type(8)))  _Float16 v8h;
typedef __attribute__((ext_vector_type(16))) _Float16 v16h;
typedef __attribute__((ext_vector_type(8)))  float    v8f;

#define NPG     28
#define GPW     16             // graphs per workgroup (WMMA M dim)
#define NODES   (GPW * NPG)    // 448
#define EDGES   (NODES * 4)    // 1792
#define KDIM    896
#define ASTRIDE 904            // padded row stride in halves (bank-conflict dodge)

// One-time per launch: fc1_w [896][256] f32 -> Bt [256][896] f16 (transposed)
__global__ void fc1w_cvt_t(const float* __restrict__ w, _Float16* __restrict__ bt) {
    int i = blockIdx.x * 256 + threadIdx.x;       // over 256*896
    if (i >= 256 * 896) return;
    int n = i / 896, k = i % 896;
    bt[i] = (_Float16)w[k * 256 + n];
}

__launch_bounds__(256, 2)
__global__ void gnn_fused(const float* __restrict__ x,
                          const int*   __restrict__ edst,   // edge_index row 1 (dst)
                          const float* __restrict__ cw,     // [2][32]
                          const float* __restrict__ cb,     // [32]
                          const _Float16* __restrict__ bt,  // [256][896] f16 (fc1_w^T)
                          const float* __restrict__ f1b,    // [256]
                          const float* __restrict__ f2w,    // [256][2]
                          const float* __restrict__ f2b,    // [2]
                          float* __restrict__ hn_out,
                          float* __restrict__ mu_out,
                          float* __restrict__ th_out) {
    __shared__ float          hsc[NODES][32];     // h * dinv[src]
    __shared__ _Float16       Ash[GPW][ASTRIDE];  // GEMM A: 16 graphs x 896 feats
    __shared__ unsigned short dstL[EDGES];
    __shared__ float          dinv[NODES];
    __shared__ float          Wc[64], Bc[32];
    __shared__ float          rowSS[GPW], rinv[GPW], x1v[GPW][2];

    const int  tid = threadIdx.x;
    const int  g0  = blockIdx.x * GPW;            // first graph of this WG
    const int  n0  = g0 * NPG;                    // first global node
    const long e0  = (long)n0 * 4;                // first global edge

    // ---- P0: stage edges + conv weights, zero reductions ----
    for (int i = tid; i < EDGES; i += 256)
        dstL[i] = (unsigned short)(edst[e0 + i] - n0);
    if (tid < 64)       Wc[tid]      = cw[tid];
    else if (tid < 96)  Bc[tid - 64] = cb[tid - 64];
    if (tid < GPW) { rowSS[tid] = 0.f; x1v[tid][0] = 0.f; x1v[tid][1] = 0.f; }
    __syncthreads();

    // ---- P1: degree -> dinv, h = x @ conv_w prescaled by dinv ----
    for (int ln = tid; ln < NODES; ln += 256) {
        int gl = ln / NPG;
        const unsigned short* de = &dstL[gl * 112];
        int cnt = 1;                               // self loop
        #pragma unroll 8
        for (int e = 0; e < 112; ++e) cnt += (de[e] == (unsigned short)ln);
        float di = rsqrtf((float)cnt);
        dinv[ln] = di;
        float x0 = x[(long)(n0 + ln) * 2 + 0];
        float x1 = x[(long)(n0 + ln) * 2 + 1];
        #pragma unroll
        for (int j = 0; j < 32; ++j)
            hsc[ln][j] = (x0 * Wc[j] + x1 * Wc[32 + j]) * di;
    }
    __syncthreads();

    // ---- P2: per-node gather, + self loop, * dinv[dst], +bias, ReLU -> f16 A ----
    for (int ln = tid; ln < NODES; ln += 256) {
        int gl = ln / NPG, ni = ln % NPG;
        float acc[32];
        #pragma unroll
        for (int j = 0; j < 32; ++j) acc[j] = hsc[ln][j];   // self-loop term
        const unsigned short* de = &dstL[gl * 112];
        for (int e = 0; e < 112; ++e) {
            if (de[e] == (unsigned short)ln) {
                int sl = gl * NPG + (e >> 2);                // src local id
                #pragma unroll
                for (int j = 0; j < 32; ++j) acc[j] += hsc[sl][j];
            }
        }
        float di = dinv[ln];
        #pragma unroll
        for (int j = 0; j < 32; ++j) {
            float v = fmaxf(acc[j] * di + Bc[j], 0.f);
            Ash[gl][ni * 32 + j] = (_Float16)v;
        }
    }
    __syncthreads();

    // ---- P3: fc1 GEMM 16 x 896 x 256 via v_wmma_f32_16x16x32_f16 ----
    const int wv   = tid >> 5;                    // wave id: owns cols [wv*32, wv*32+32)
    const int lane = tid & 31;
    const int lm   = lane & 15;
    const int off  = (lane < 16) ? 0 : 8;         // K sub-block per half-wave (ISA layout)
    const int base0 = wv * 32, base1 = base0 + 16;

    v8f c0 = {}, c1 = {};
    const _Float16* btA  = bt + (long)(base0 + lm) * KDIM;
    const _Float16* btB  = bt + (long)(base1 + lm) * KDIM;
    const _Float16* arow = &Ash[lm][0];

    for (int kb = 0; kb < 28; ++kb) {
        int k0 = kb * 32 + off;
        __builtin_prefetch(btA + k0 + 64, 0, 1);
        __builtin_prefetch(btB + k0 + 64, 0, 1);
        v8h alo = *(const v8h*)(arow + k0);
        v8h ahi = *(const v8h*)(arow + k0 + 16);
        v16h a  = __builtin_shufflevector(alo, ahi, 0,1,2,3,4,5,6,7,8,9,10,11,12,13,14,15);
        v8h b0l = *(const v8h*)(btA + k0);
        v8h b0h = *(const v8h*)(btA + k0 + 16);
        v16h b0 = __builtin_shufflevector(b0l, b0h, 0,1,2,3,4,5,6,7,8,9,10,11,12,13,14,15);
        v8h b1l = *(const v8h*)(btB + k0);
        v8h b1h = *(const v8h*)(btB + k0 + 16);
        v16h b1 = __builtin_shufflevector(b1l, b1h, 0,1,2,3,4,5,6,7,8,9,10,11,12,13,14,15);
        c0 = __builtin_amdgcn_wmma_f32_16x16x32_f16(false, a, false, b0, (short)0, c0, false, false);
        c1 = __builtin_amdgcn_wmma_f32_16x16x32_f16(false, a, false, b1, (short)0, c1, false, false);
    }

    // ---- P4: +fc1 bias, row sum of squares ----
    const int colA = base0 + lm, colB = base1 + lm;
    float bA = f1b[colA], bB = f1b[colB];
    #pragma unroll
    for (int r = 0; r < 8; ++r) { c0[r] += bA; c1[r] += bB; }
    #pragma unroll
    for (int r = 0; r < 8; ++r) {
        int row = r + ((lane < 16) ? 0 : 8);
        atomicAdd(&rowSS[row], c0[r] * c0[r] + c1[r] * c1[r]);   // ds_add_f32
    }
    __syncthreads();
    if (tid < GPW) rinv[tid] = 1.0f / fmaxf(sqrtf(rowSS[tid]), 1e-12f);
    __syncthreads();

    // ---- P5: normalize, store hn, fc2 partial dots ----
    float w0a = f2w[colA * 2], w1a = f2w[colA * 2 + 1];
    float w0b = f2w[colB * 2], w1b = f2w[colB * 2 + 1];
    #pragma unroll
    for (int r = 0; r < 8; ++r) {
        int row = r + ((lane < 16) ? 0 : 8);
        float ri = rinv[row];
        float a0 = c0[r] * ri, a1 = c1[r] * ri;
        long  gr = (long)(g0 + row);
        hn_out[gr * 256 + colA] = a0;
        hn_out[gr * 256 + colB] = a1;
        atomicAdd(&x1v[row][0], a0 * w0a + a1 * w0b);
        atomicAdd(&x1v[row][1], a0 * w1a + a1 * w1b);
    }
    __syncthreads();

    // ---- P6: heads ----
    if (tid < GPW) {
        float v0 = x1v[tid][0] + f2b[0];
        float v1 = x1v[tid][1] + f2b[1];
        mu_out[g0 + tid] = 1.0f / (1.0f + __expf(-v0));                       // sigmoid
        th_out[g0 + tid] = fmaxf(v1, 0.f) + log1pf(__expf(-fabsf(v1)));      // softplus
    }
}

extern "C" void kernel_launch(void* const* d_in, const int* in_sizes, int n_in,
                              void* d_out, int out_size, void* d_ws, size_t ws_size,
                              hipStream_t stream) {
    const float* x   = (const float*)d_in[0];
    const int*   ei  = (const int*)  d_in[1];   // [2, E] flat: src then dst
    const float* cw  = (const float*)d_in[2];
    const float* cb  = (const float*)d_in[3];
    const float* f1w = (const float*)d_in[4];
    const float* f1b = (const float*)d_in[5];
    const float* f2w = (const float*)d_in[6];
    const float* f2b = (const float*)d_in[7];

    const int  N = in_sizes[0] / 2;
    const int  B = N / NPG;
    const long E = (long)N * 4;

    _Float16* bt = (_Float16*)d_ws;             // 256*896*2 = 458752 bytes
    fc1w_cvt_t<<<(256 * 896 + 255) / 256, 256, 0, stream>>>(f1w, bt);

    float* hn = (float*)d_out;
    float* mu = hn + (long)B * 256;
    float* th = mu + B;
    gnn_fused<<<B / GPW, 256, 0, stream>>>(x, ei + E, cw, cb, bt, f1b, f2w, f2b,
                                           hn, mu, th);
}
